// ResidualBlock_49838800503534
// MI455X (gfx1250) — compile-verified, matrix operand order checked by reading, not run on hardware
//
#include <hip/hip_runtime.h>
#include <hip/hip_bf16.h>
#include <math.h>

// ---------------- problem constants ----------------
#define D_MODEL   64
#define D_STATE   16
#define D_CONV    4
#define NUM_NODES 325
#define HISL      144
#define D_INNER   128            // 2 * 64
#define NT        (NUM_NODES * HISL)   // 46800 tokens
#define MT16      (NT / 16)            // 2925 row tiles
#define MT64      ((MT16 + 3) / 4)     // 732 blocks of 4 row-tiles
#define DBC       (1 + 2 * D_STATE)    // 33

typedef __attribute__((ext_vector_type(16))) _Float16 v16h;
typedef __attribute__((ext_vector_type(8)))  float    v8f;

// ---------------- helpers ----------------
__device__ __forceinline__ float sigmoidf_(float x) { return 1.0f / (1.0f + __expf(-x)); }
__device__ __forceinline__ float siluf_(float x)    { return x * sigmoidf_(x); }

// Load 16 f16 values for one lane of a 16x32 (f16) WMMA operand.
// Per ISA 7.12.2: lane half h holds K = [8h, 8h+8) and [16+8h, 16+8h+8).
// p points at (row base + k0 + 8h); fp32 source, converted to f16.
__device__ __forceinline__ v16h load16h(const float* __restrict__ p) {
    const float4 f0 = *reinterpret_cast<const float4*>(p);
    const float4 f1 = *reinterpret_cast<const float4*>(p + 4);
    const float4 f2 = *reinterpret_cast<const float4*>(p + 16);
    const float4 f3 = *reinterpret_cast<const float4*>(p + 20);
    v16h r;
    r[0]  = (_Float16)f0.x; r[1]  = (_Float16)f0.y; r[2]  = (_Float16)f0.z; r[3]  = (_Float16)f0.w;
    r[4]  = (_Float16)f1.x; r[5]  = (_Float16)f1.y; r[6]  = (_Float16)f1.z; r[7]  = (_Float16)f1.w;
    r[8]  = (_Float16)f2.x; r[9]  = (_Float16)f2.y; r[10] = (_Float16)f2.z; r[11] = (_Float16)f2.w;
    r[12] = (_Float16)f3.x; r[13] = (_Float16)f3.y; r[14] = (_Float16)f3.z; r[15] = (_Float16)f3.w;
    return r;
}

// ---------------- GEMM: out[M,N] = A[M,K] @ W[N,K]^T (+bias, +relu) ----------------
// block = 128 threads (4 waves). Each wave computes a 16x64 strip:
// one A fragment per k-step reused across 4 WMMA (4 N tiles) -> 4x less A traffic.
// gridDim.x = ceil(M/64)  (wave wv handles row tile blockIdx.x*4+wv, whole-wave guard),
// gridDim.y = N/64.  K templated (64/128) so the k-loop fully unrolls.
// MODE: 0 none, 1 +bias, 2 +bias+relu.
template <int K, int MODE>
__global__ __launch_bounds__(128)
void gemm_wmma_f16(const float* __restrict__ A, const float* __restrict__ W,
                   const float* __restrict__ bias, float* __restrict__ out,
                   int M, int N) {
    const int lane = threadIdx.x & 31;
    const int wv   = threadIdx.x >> 5;
    const int mt   = blockIdx.x * 4 + wv;        // row tile for this wave
    if (mt * 16 >= M) return;                    // whole-wave exit: EXEC stays all-1s

    const int hh   = lane >> 4;                  // lane half
    const int sub  = lane & 15;
    const int row  = mt * 16 + sub;
    const int ncol0 = blockIdx.y * 64;           // 4 N tiles: ncol0 + 16*j

    const float* __restrict__ arow = A + (size_t)row * K;
    const float* __restrict__ brow0 = W + (size_t)(ncol0 + sub) * K;

    v8f acc0 = {}, acc1 = {}, acc2 = {}, acc3 = {};
#pragma unroll
    for (int k0 = 0; k0 < K; k0 += 32) {
        const v16h a  = load16h(arow + k0 + 8 * hh);
        const v16h b0 = load16h(brow0 + 0 * 16 * K + k0 + 8 * hh);
        const v16h b1 = load16h(brow0 + 1 * 16 * K + k0 + 8 * hh);
        const v16h b2 = load16h(brow0 + 2 * 16 * K + k0 + 8 * hh);
        const v16h b3 = load16h(brow0 + 3 * 16 * K + k0 + 8 * hh);
        acc0 = __builtin_amdgcn_wmma_f32_16x16x32_f16(false, a, false, b0, (short)0, acc0, false, false);
        acc1 = __builtin_amdgcn_wmma_f32_16x16x32_f16(false, a, false, b1, (short)0, acc1, false, false);
        acc2 = __builtin_amdgcn_wmma_f32_16x16x32_f16(false, a, false, b2, (short)0, acc2, false, false);
        acc3 = __builtin_amdgcn_wmma_f32_16x16x32_f16(false, a, false, b3, (short)0, acc3, false, false);
    }

    // D layout: VGPR r -> M = 8*hh + r, N = lane&15
    const int mbase = mt * 16 + hh * 8;
    v8f accs[4] = {acc0, acc1, acc2, acc3};
#pragma unroll
    for (int j = 0; j < 4; ++j) {
        const int n = ncol0 + j * 16 + sub;
        const float bv = (MODE >= 1) ? bias[n] : 0.0f;
#pragma unroll
        for (int r = 0; r < 8; ++r) {
            float v = accs[j][r] + bv;
            if (MODE == 2) v = fmaxf(v, 0.0f);
            out[(size_t)(mbase + r) * N + n] = v;
        }
    }
}

// ---------------- RMSNorm + token-order permute ----------------
// mode 0: out token ts=(l*325+n) <- in token (n*144+l)   [temporal->spatial]
// mode 1: out token tt=(n*144+l) <- in token (l*325+n)   [spatial->temporal]
__global__ __launch_bounds__(64)
void rms_permute(const float* __restrict__ in, float* __restrict__ out,
                 const float* __restrict__ w, int mode) {
    const int t = blockIdx.x;
    const int d = threadIdx.x;
    int in_tok;
    if (mode == 0) { int l = t / NUM_NODES, n = t % NUM_NODES; in_tok = n * HISL + l; }
    else           { int n = t / HISL,      l = t % HISL;      in_tok = l * NUM_NODES + n; }
    float v = in[(size_t)in_tok * D_MODEL + d];
    __shared__ float s[64];
    s[d] = v * v;
    __syncthreads();
#pragma unroll
    for (int off = 32; off > 0; off >>= 1) {
        if (d < off) s[d] += s[d + off];
        __syncthreads();
    }
    float scale = rsqrtf(s[0] * (1.0f / D_MODEL) + 1e-5f);
    out[(size_t)t * D_MODEL + d] = v * scale * w[d];
}

// ---------------- residual add + LayerNorm ----------------
__global__ __launch_bounds__(64)
void add_layernorm(const float* __restrict__ a, const float* __restrict__ b,
                   const float* __restrict__ w, const float* __restrict__ bi,
                   float* __restrict__ out) {
    const int t = blockIdx.x;
    const int d = threadIdx.x;
    const size_t idx = (size_t)t * D_MODEL + d;
    float v = a[idx] + b[idx];
    __shared__ float s[64];
    s[d] = v;
    __syncthreads();
#pragma unroll
    for (int off = 32; off > 0; off >>= 1) {
        if (d < off) s[d] += s[d + off];
        __syncthreads();
    }
    float mu = s[0] * (1.0f / D_MODEL);
    __syncthreads();
    float dv = v - mu;
    s[d] = dv * dv;
    __syncthreads();
#pragma unroll
    for (int off = 32; off > 0; off >>= 1) {
        if (d < off) s[d] += s[d + off];
        __syncthreads();
    }
    float var = s[0] * (1.0f / D_MODEL);
    out[idx] = dv * rsqrtf(var + 1e-5f) * w[d] + bi[d];
}

// ---------------- Mamba core: conv4 + x_proj + selective scan + gate ----------------
// One workgroup (128 threads == D_INNER channels) per sequence; tokens contiguous.
// xz row = [ix(0..127) | z(128..255)]. Writes gated y[NT, 128].
__global__ __launch_bounds__(128)
void mamba_core(const float* __restrict__ xz, float* __restrict__ yg,
                const float* __restrict__ conv_w, const float* __restrict__ conv_b,
                const float* __restrict__ xw,     const float* __restrict__ dtw,
                const float* __restrict__ dtb,    const float* __restrict__ A_log,
                const float* __restrict__ Dv,     int L) {
    const int c = threadIdx.x;
    const long base = (long)blockIdx.x * L;

    __shared__ float s_xw[DBC * D_INNER];   // x_proj weights (33 x 128)
    __shared__ float s_u[D_INNER];
    __shared__ float s_dbc[DBC];
    for (int i = c; i < DBC * D_INNER; i += D_INNER) s_xw[i] = xw[i];

    float Ac[D_STATE], hst[D_STATE];
#pragma unroll
    for (int j = 0; j < D_STATE; ++j) {
        Ac[j]  = -__expf(A_log[c * D_STATE + j]);
        hst[j] = 0.0f;
    }
    const float cw0 = conv_w[c * 4 + 0], cw1 = conv_w[c * 4 + 1];
    const float cw2 = conv_w[c * 4 + 2], cw3 = conv_w[c * 4 + 3];
    const float cb  = conv_b[c];
    const float dw  = dtw[c], db = dtb[c], Dc = Dv[c];

    // conv window over ix: positions t-2, t-1, t, t+1
    float x0 = 0.0f, x1 = 0.0f;
    float x2 = xz[base * 256 + c];
    float x3 = (L > 1) ? xz[(base + 1) * 256 + c] : 0.0f;
    __syncthreads();   // s_xw ready

    for (int t = 0; t < L; ++t) {
        float up = cw0 * x0 + cw1 * x1 + cw2 * x2 + cw3 * x3 + cb;
        float u  = siluf_(up);
        s_u[c] = u;
        __syncthreads();
        if (c < DBC) {
            float acc = 0.0f;
            const float* wr = &s_xw[c * D_INNER];
#pragma unroll 8
            for (int k = 0; k < D_INNER; ++k) acc += wr[k] * s_u[k];
            s_dbc[c] = acc;
        }
        __syncthreads();
        float dtv   = s_dbc[0] * dw + db;
        float delta = (dtv > 20.0f) ? dtv : log1pf(__expf(dtv));
        float du    = delta * u;
        float y     = 0.0f;
#pragma unroll
        for (int j = 0; j < D_STATE; ++j) {
            float dA = __expf(delta * Ac[j]);
            hst[j] = fmaf(dA, hst[j], du * s_dbc[1 + j]);
            y = fmaf(hst[j], s_dbc[1 + D_STATE + j], y);
        }
        y = fmaf(Dc, u, y);
        float zv = xz[(base + t) * 256 + 128 + c];
        yg[(base + t) * 128 + c] = y * siluf_(zv);
        // slide window
        x0 = x1; x1 = x2; x2 = x3;
        x3 = (t + 2 < L) ? xz[(base + t + 2) * 256 + c] : 0.0f;
        // next iteration's first barrier orders s_dbc reads vs. its rewrite
    }
}

// ---------------- host-side orchestration ----------------
extern "C" void kernel_launch(void* const* d_in, const int* in_sizes, int n_in,
                              void* d_out, int out_size, void* d_ws, size_t ws_size,
                              hipStream_t stream) {
    // flattened input order (setup_inputs insertion order; nested dicts in-order)
    const float* x       = (const float*)d_in[0];
    // p_mixer (temporal)
    const float* t_inw   = (const float*)d_in[1];
    const float* t_cw    = (const float*)d_in[2];
    const float* t_cb    = (const float*)d_in[3];
    const float* t_xw    = (const float*)d_in[4];
    const float* t_dtw   = (const float*)d_in[5];
    const float* t_dtb   = (const float*)d_in[6];
    const float* t_alog  = (const float*)d_in[7];
    const float* t_D     = (const float*)d_in[8];
    const float* t_outw  = (const float*)d_in[9];
    // p_mixer1 (spatial)
    const float* s_inw   = (const float*)d_in[10];
    const float* s_cw    = (const float*)d_in[11];
    const float* s_cb    = (const float*)d_in[12];
    const float* s_xw    = (const float*)d_in[13];
    const float* s_dtw   = (const float*)d_in[14];
    const float* s_dtb   = (const float*)d_in[15];
    const float* s_alog  = (const float*)d_in[16];
    const float* s_D     = (const float*)d_in[17];
    const float* s_outw  = (const float*)d_in[18];
    const float* norm_w  = (const float*)d_in[19];
    const float* ff_w1   = (const float*)d_in[20];
    const float* ff_b1   = (const float*)d_in[21];
    const float* ff_w2   = (const float*)d_in[22];
    const float* ff_b2   = (const float*)d_in[23];
    const float* ln1_w   = (const float*)d_in[24];
    const float* ln1_b   = (const float*)d_in[25];
    const float* ln2_w   = (const float*)d_in[26];
    const float* ln2_b   = (const float*)d_in[27];

    // workspace layout (bytes, 256-aligned)
    char* ws = (char*)d_ws;
    float* normB = (float*)(ws);                                     // NT*64   (12 MB)
    float* xzB   = (float*)(ws + (size_t)NT * 64 * 4);               // NT*256  (48 MB)
    float* ygB   = (float*)(ws + (size_t)NT * (64 + 256) * 4);       // NT*128  (24 MB)
    float* stB   = (float*)(ws + (size_t)NT * (64 + 256 + 128) * 4); // NT*64   (12 MB)

    dim3 blk128(128), blk64(64);

    // ---- spatial mixer (p_mixer1), seq over nodes (len 325), 144 sequences ----
    rms_permute<<<NT, blk64, 0, stream>>>(x, normB, norm_w, 0);
    gemm_wmma_f16<64, 0><<<dim3(MT64, 4), blk128, 0, stream>>>(normB, s_inw, nullptr, xzB,
                                                               NT, 2 * D_INNER);
    mamba_core<<<HISL, blk128, 0, stream>>>(xzB, ygB, s_cw, s_cb, s_xw, s_dtw, s_dtb,
                                            s_alog, s_D, NUM_NODES);
    gemm_wmma_f16<128, 0><<<dim3(MT64, 1), blk128, 0, stream>>>(ygB, s_outw, nullptr, stB,
                                                                NT, D_MODEL);

    // ---- temporal mixer (p_mixer), seq over time (len 144), 325 sequences ----
    rms_permute<<<NT, blk64, 0, stream>>>(stB, normB, norm_w, 1);
    gemm_wmma_f16<64, 0><<<dim3(MT64, 4), blk128, 0, stream>>>(normB, t_inw, nullptr, xzB,
                                                               NT, 2 * D_INNER);
    mamba_core<<<NUM_NODES, blk128, 0, stream>>>(xzB, ygB, t_cw, t_cb, t_xw, t_dtw, t_dtb,
                                                 t_alog, t_D, HISL);
    gemm_wmma_f16<128, 0><<<dim3(MT64, 1), blk128, 0, stream>>>(ygB, t_outw, nullptr, stB,
                                                                NT, D_MODEL);

    // ---- residual + LN1, FFN (WMMA), residual + LN2 ----
    add_layernorm<<<NT, blk64, 0, stream>>>(x, stB, ln1_w, ln1_b, normB);      // h
    gemm_wmma_f16<64, 2><<<dim3(MT64, 2), blk128, 0, stream>>>(normB, ff_w1, ff_b1, ygB,
                                                               NT, 2 * D_MODEL);   // relu
    gemm_wmma_f16<128, 1><<<dim3(MT64, 1), blk128, 0, stream>>>(ygB, ff_w2, ff_b2, xzB,
                                                                NT, D_MODEL);
    add_layernorm<<<NT, blk64, 0, stream>>>(normB, xzB, ln2_w, ln2_b, (float*)d_out);
}
